// Conv2d_69535520522666
// MI455X (gfx1250) — matrix operand order repaired and checked by reading, not compiled, not run on hardware
//
#include <hip/hip_runtime.h>

// ---------------------------------------------------------------------------
// Implicit-GEMM 3x3 conv (NCHW, pad 1, stride 1), bf16 WMMA with f32 accum.
// x[1,256,512,512] f32, w[256,256,3,3] f32 -> out[1,256,512,512] f32
//
// Pass 1: x NCHW f32 -> xt NHWC bf16 (128 MiB, fits in 192 MB L2)
//         w OIHW f32 -> wt [tap][oc][ic] bf16
// Pass 2: WMMA main loop; LDS staging via GLOBAL_LOAD_ASYNC_TO_LDS_B128
//         (ASYNCcnt) when available, plain 16B copies otherwise.
// Wave tile 32oc x 64px (8 accumulators) => 1.5 ds_load_b128 per WMMA.
// ---------------------------------------------------------------------------

typedef __attribute__((ext_vector_type(16))) __bf16 v16bf;
typedef __attribute__((ext_vector_type(8)))  float v8f;
typedef __attribute__((__vector_size__(16))) int v4i_t;

constexpr int Hn = 512, Wn = 512;
constexpr int Cin = 256, Cout = 256;
constexpr int HW = Hn * Wn;

constexpr int TILE_OC = 64;          // output channels per block
constexpr int TILE_H  = 4;           // output rows per block
constexpr int TILE_W  = 64;          // output cols per block
constexpr int ICCH    = 32;          // input-channel chunk (WMMA K)
constexpr int IN_ROWS = TILE_H + 2;  // 6 (halo)
constexpr int IN_COLS = TILE_W + 2;  // 66 (halo)
constexpr int ICP     = 40;          // padded ic stride (ushorts) = 80B

#define ASG __attribute__((address_space(1)))
#define ASL __attribute__((address_space(3)))

#if defined(__has_builtin)
#if __has_builtin(__builtin_amdgcn_global_load_async_to_lds_b128)
#define HAVE_ASYNC_LDS 1
#endif
#if __has_builtin(__builtin_amdgcn_s_wait_asynccnt)
#define HAVE_WAIT_ASYNC 1
#endif
#endif

__device__ __forceinline__ void cp16_g2l(const void* g, void* l) {
#if defined(HAVE_ASYNC_LDS)
  __builtin_amdgcn_global_load_async_to_lds_b128(
      (ASG v4i_t*)const_cast<void*>(g), (ASL v4i_t*)l, 0, 0);
#else
  *(uint4*)l = *(const uint4*)g;
#endif
}

__device__ __forceinline__ void wait_async_lds() {
#if defined(HAVE_ASYNC_LDS)
#if defined(HAVE_WAIT_ASYNC)
  __builtin_amdgcn_s_wait_asynccnt(0);
#else
  asm volatile("s_wait_asynccnt 0" ::: "memory");
#endif
#endif
}

__device__ __forceinline__ unsigned short f2bf(float f) {
  // round-to-nearest-even fp32 -> bf16
  unsigned u = __float_as_uint(f);
  u += 0x7FFFu + ((u >> 16) & 1u);
  return (unsigned short)(u >> 16);
}

// A fragment (16x32 bf16): lanes 0-15 hold K0-15, lanes 16-31 hold K16-31
__device__ __forceinline__ v16bf load_a(const unsigned short* row, int halfSel) {
  const char* p = (const char*)row;
  union { uint4 q[2]; v16bf v; } u;
  u.q[0] = *(const uint4*)(p + halfSel * 32);
  u.q[1] = *(const uint4*)(p + 16 + halfSel * 32);
  return u.v;
}

// B fragment (32x16 bf16): lanes 0-15 hold K0-15, lanes 16-31 hold K16-31
__device__ __forceinline__ v16bf load_b(const unsigned short* row, int halfSel) {
  const char* p = (const char*)row;
  union { uint4 q[2]; v16bf v; } u;
  u.q[0] = *(const uint4*)(p + halfSel * 32);
  u.q[1] = *(const uint4*)(p + 16 + halfSel * 32);
  return u.v;
}

__device__ __forceinline__ v8f wmma_bf16(v16bf a, v16bf b, v8f c) {
  return __builtin_amdgcn_wmma_f32_16x16x32_bf16(false, a, false, b,
                                                 (short)0, c, false, false);
}

// ---------------------------------------------------------------------------
// Pass 1a: x NCHW f32 -> xt NHWC bf16 (coalesced both sides via LDS transpose)
// ---------------------------------------------------------------------------
constexpr int XP = 264;  // padded ushort stride per pixel (528B, 16B multiple)

__global__ __launch_bounds__(256)
void transform_x_kernel(const float* __restrict__ x,
                        unsigned short* __restrict__ xt) {
  __shared__ unsigned short t[64 * XP];  // 33792 B
  const int tid  = threadIdx.x;
  const int w0   = blockIdx.x * 64;
  const int h    = blockIdx.y;
  const int wloc = tid & 63;
  const int icq  = tid >> 6;  // 0..3
#pragma unroll 4
  for (int icg = 0; icg < 64; ++icg) {
    int ic = icg * 4 + icq;
    float v = x[(size_t)ic * HW + (size_t)h * Wn + w0 + wloc];
    t[wloc * XP + ic] = f2bf(v);
  }
  __syncthreads();
  // write 64 pixels x 512B, fully coalesced
  for (int e = tid; e < 64 * 32; e += 256) {
    int p = e >> 5, sub = e & 31;
    uint4 val = *(const uint4*)((const char*)t + p * (XP * 2) + sub * 16);
    *(uint4*)((char*)xt + ((size_t)h * Wn + w0 + p) * (Cin * 2) + sub * 16) = val;
  }
}

// ---------------------------------------------------------------------------
// Pass 1b: w OIHW f32 -> wt [tap][oc][ic] bf16
// ---------------------------------------------------------------------------
__global__ __launch_bounds__(256)
void transform_w_kernel(const float* __restrict__ wgt,
                        unsigned short* __restrict__ wt) {
  int idx = blockIdx.x * 256 + threadIdx.x;  // oc*256 + ic, 65536 total
  int oc = idx >> 8, ic = idx & 255;
  const float* src = wgt + (size_t)idx * 9;
#pragma unroll
  for (int tap = 0; tap < 9; ++tap)
    wt[((size_t)tap * Cout + oc) * Cin + ic] = f2bf(src[tap]);
}

// ---------------------------------------------------------------------------
// Pass 2: main WMMA kernel reading bf16 xt / wt
// ---------------------------------------------------------------------------
__global__ __launch_bounds__(256)
void conv3x3_wmma_fast(const unsigned short* __restrict__ xt,
                       const unsigned short* __restrict__ wt,
                       float* __restrict__ out) {
  __shared__ unsigned short s_in[IN_ROWS * IN_COLS * ICP];  // 31680 B
  __shared__ unsigned short s_w[2 * 3 * TILE_OC * ICP];     // 30720 B (dbl buf)

  const int tid     = threadIdx.x;
  const int lane    = tid & 31;
  const int wave    = tid >> 5;
  const int wave_r  = wave & 3;
  const int wave_oc = wave >> 2;

  const int w0  = blockIdx.x * TILE_W;
  const int h0  = blockIdx.y * TILE_H;
  const int oc0 = blockIdx.z * TILE_OC;

  const int m       = lane & 15;
  const int halfSel = lane >> 4;

  const v8f vzero = {0.f, 0.f, 0.f, 0.f, 0.f, 0.f, 0.f, 0.f};
  v8f acc[2][4];
#pragma unroll
  for (int i = 0; i < 2; ++i)
#pragma unroll
    for (int j = 0; j < 4; ++j) acc[i][j] = vzero;

  for (int chunk = 0; chunk < Cin / ICCH; ++chunk) {
    // ---- stage input halo tile: 16B async copies, zero-fill halo OOB ----
    for (int e = tid; e < IN_ROWS * IN_COLS * 4; e += 256) {
      int sub = e & 3;
      int pc  = e >> 2;
      int col = pc % IN_COLS;
      int row = pc / IN_COLS;
      int gh  = h0 - 1 + row;
      int gw  = w0 - 1 + col;
      char* lb = (char*)&s_in[(row * IN_COLS + col) * ICP] + sub * 16;
      if ((unsigned)gh < (unsigned)Hn && (unsigned)gw < (unsigned)Wn) {
        const char* g = (const char*)xt +
            (((size_t)gh * Wn + gw) * Cin + chunk * ICCH) * 2 + sub * 16;
        cp16_g2l(g, lb);
      } else {
        *(uint4*)lb = make_uint4(0u, 0u, 0u, 0u);
      }
    }
    // ---- stage kh=0 weight row into buffer 0 ----
    for (int e = tid; e < 3 * TILE_OC * 4; e += 256) {
      int sub = e & 3;
      int rc  = e >> 2;
      int oc  = rc & 63;
      int dw  = rc >> 6;
      char* lb = (char*)&s_w[(dw * TILE_OC + oc) * ICP] + sub * 16;
      const char* g = (const char*)wt +
          (((size_t)dw * Cout + oc0 + oc) * Cin + chunk * ICCH) * 2 + sub * 16;
      cp16_g2l(g, lb);
    }
    wait_async_lds();
    __syncthreads();

#pragma unroll
    for (int dh = 0; dh < 3; ++dh) {
      const int buf = dh & 1;
      // prefetch next kh weight row into the other buffer while computing
      if (dh < 2) {
        const int nbuf = buf ^ 1;
        for (int e = tid; e < 3 * TILE_OC * 4; e += 256) {
          int sub = e & 3;
          int rc  = e >> 2;
          int oc  = rc & 63;
          int dw  = rc >> 6;
          char* lb = (char*)&s_w[((nbuf * 3 + dw) * TILE_OC + oc) * ICP] + sub * 16;
          const char* g = (const char*)wt +
              (((size_t)((dh + 1) * 3 + dw) * Cout + oc0 + oc) * Cin +
               chunk * ICCH) * 2 + sub * 16;
          cp16_g2l(g, lb);
        }
      }
#pragma unroll
      for (int dw = 0; dw < 3; ++dw) {
        const unsigned short* wrow =
            &s_w[((buf * 3 + dw) * TILE_OC + wave_oc * 32) * ICP];
        v16bf a0 = load_a(wrow + m * ICP, halfSel);
        v16bf a1 = load_a(wrow + (16 + m) * ICP, halfSel);
        const unsigned short* irow =
            &s_in[((wave_r + dh) * IN_COLS + dw) * ICP];
        v16bf b0 = load_b(irow + (0 * 16 + m) * ICP, halfSel);
        v16bf b1 = load_b(irow + (1 * 16 + m) * ICP, halfSel);
        v16bf b2 = load_b(irow + (2 * 16 + m) * ICP, halfSel);
        v16bf b3 = load_b(irow + (3 * 16 + m) * ICP, halfSel);
        acc[0][0] = wmma_bf16(a0, b0, acc[0][0]);
        acc[0][1] = wmma_bf16(a0, b1, acc[0][1]);
        acc[0][2] = wmma_bf16(a0, b2, acc[0][2]);
        acc[0][3] = wmma_bf16(a0, b3, acc[0][3]);
        acc[1][0] = wmma_bf16(a1, b0, acc[1][0]);
        acc[1][1] = wmma_bf16(a1, b1, acc[1][1]);
        acc[1][2] = wmma_bf16(a1, b2, acc[1][2]);
        acc[1][3] = wmma_bf16(a1, b3, acc[1][3]);
      }
      wait_async_lds();
      __syncthreads();
    }
  }

  // ---- scatter accumulators: VGPR i -> M = i + 8*halfSel, N = lane&15 ----
  const int h   = h0 + wave_r;
  const int ocb = oc0 + wave_oc * 32;
  const int n   = lane & 15;
  const size_t rowoff = (size_t)h * Wn + w0;
#pragma unroll
  for (int i = 0; i < 8; ++i) {
    const int mrow = i + halfSel * 8;
#pragma unroll
    for (int j = 0; j < 4; ++j) {
      out[(size_t)(ocb + mrow) * HW + rowoff + j * 16 + n]      = acc[0][j][i];
      out[(size_t)(ocb + 16 + mrow) * HW + rowoff + j * 16 + n] = acc[1][j][i];
    }
  }
}

// ---------------------------------------------------------------------------
// Fallback (round-2 kernel): direct fp32 reads, used if workspace too small.
// ---------------------------------------------------------------------------
__global__ __launch_bounds__(256)
void conv3x3_wmma_fallback(const float* __restrict__ x,
                           const float* __restrict__ wgt,
                           float* __restrict__ out) {
  __shared__ unsigned short s_in[IN_ROWS * IN_COLS * ICP];
  __shared__ unsigned short s_w[2 * 3 * TILE_OC * ICP];

  const int tid     = threadIdx.x;
  const int lane    = tid & 31;
  const int wave    = tid >> 5;
  const int wave_r  = wave & 3;
  const int wave_oc = wave >> 2;

  const int w0  = blockIdx.x * TILE_W;
  const int h0  = blockIdx.y * TILE_H;
  const int oc0 = blockIdx.z * TILE_OC;

  const int m       = lane & 15;
  const int halfSel = lane >> 4;

  const v8f vzero = {0.f, 0.f, 0.f, 0.f, 0.f, 0.f, 0.f, 0.f};
  v8f acc[2][4];
#pragma unroll
  for (int i = 0; i < 2; ++i)
#pragma unroll
    for (int j = 0; j < 4; ++j) acc[i][j] = vzero;

  for (int chunk = 0; chunk < Cin / ICCH; ++chunk) {
    for (int e = tid; e < IN_ROWS * IN_COLS * ICCH; e += 256) {
      int col = e % IN_COLS;
      int t   = e / IN_COLS;
      int row = t % IN_ROWS;
      int ic  = t / IN_ROWS;
      int gh  = h0 - 1 + row;
      int gw  = w0 - 1 + col;
      float v = 0.0f;
      if ((unsigned)gh < (unsigned)Hn && (unsigned)gw < (unsigned)Wn)
        v = x[(size_t)(chunk * ICCH + ic) * HW + gh * Wn + gw];
      s_in[(row * IN_COLS + col) * ICP + ic] = f2bf(v);
    }
    for (int e = tid; e < 3 * TILE_OC * ICCH; e += 256) {
      int dw = e % 3;
      int t  = e / 3;
      int ic = t % ICCH;
      int oc = t / ICCH;
      float v = wgt[((size_t)(oc0 + oc) * Cin + chunk * ICCH + ic) * 9 + dw];
      s_w[(dw * TILE_OC + oc) * ICP + ic] = f2bf(v);
    }
    __syncthreads();

#pragma unroll
    for (int dh = 0; dh < 3; ++dh) {
      const int buf = dh & 1;
      if (dh < 2) {
        const int nbuf = buf ^ 1;
        for (int e = tid; e < 3 * TILE_OC * ICCH; e += 256) {
          int dw = e % 3;
          int t  = e / 3;
          int ic = t % ICCH;
          int oc = t / ICCH;
          float v = wgt[((size_t)(oc0 + oc) * Cin + chunk * ICCH + ic) * 9 +
                        (dh + 1) * 3 + dw];
          s_w[((nbuf * 3 + dw) * TILE_OC + oc) * ICP + ic] = f2bf(v);
        }
      }
#pragma unroll
      for (int dw = 0; dw < 3; ++dw) {
        const unsigned short* wrow =
            &s_w[((buf * 3 + dw) * TILE_OC + wave_oc * 32) * ICP];
        v16bf a0 = load_a(wrow + m * ICP, halfSel);
        v16bf a1 = load_a(wrow + (16 + m) * ICP, halfSel);
        const unsigned short* irow =
            &s_in[((wave_r + dh) * IN_COLS + dw) * ICP];
        v16bf b0 = load_b(irow + (0 * 16 + m) * ICP, halfSel);
        v16bf b1 = load_b(irow + (1 * 16 + m) * ICP, halfSel);
        v16bf b2 = load_b(irow + (2 * 16 + m) * ICP, halfSel);
        v16bf b3 = load_b(irow + (3 * 16 + m) * ICP, halfSel);
        acc[0][0] = wmma_bf16(a0, b0, acc[0][0]);
        acc[0][1] = wmma_bf16(a0, b1, acc[0][1]);
        acc[0][2] = wmma_bf16(a0, b2, acc[0][2]);
        acc[0][3] = wmma_bf16(a0, b3, acc[0][3]);
        acc[1][0] = wmma_bf16(a1, b0, acc[1][0]);
        acc[1][1] = wmma_bf16(a1, b1, acc[1][1]);
        acc[1][2] = wmma_bf16(a1, b2, acc[1][2]);
        acc[1][3] = wmma_bf16(a1, b3, acc[1][3]);
      }
      __syncthreads();
    }
  }

  const int h   = h0 + wave_r;
  const int ocb = oc0 + wave_oc * 32;
  const int n   = lane & 15;
  const size_t rowoff = (size_t)h * Wn + w0;
#pragma unroll
  for (int i = 0; i < 8; ++i) {
    const int mrow = i + halfSel * 8;
#pragma unroll
    for (int j = 0; j < 4; ++j) {
      out[(size_t)(ocb + mrow) * HW + rowoff + j * 16 + n]      = acc[0][j][i];
      out[(size_t)(ocb + 16 + mrow) * HW + rowoff + j * 16 + n] = acc[1][j][i];
    }
  }
}

extern "C" void kernel_launch(void* const* d_in, const int* in_sizes, int n_in,
                              void* d_out, int out_size, void* d_ws, size_t ws_size,
                              hipStream_t stream) {
  const float* x   = (const float*)d_in[0];
  const float* wgt = (const float*)d_in[1];
  float* out       = (float*)d_out;

  const size_t xt_bytes = (size_t)HW * Cin * 2;        // 128 MiB
  const size_t wt_bytes = (size_t)9 * Cout * Cin * 2;  // 1.125 MiB
  dim3 grid(Wn / TILE_W, Hn / TILE_H, Cout / TILE_OC); // 8 x 128 x 4

  if (ws_size >= xt_bytes + wt_bytes) {
    unsigned short* xt = (unsigned short*)d_ws;
    unsigned short* wt = (unsigned short*)((char*)d_ws + xt_bytes);
    transform_x_kernel<<<dim3(Wn / 64, Hn), 256, 0, stream>>>(x, xt);
    transform_w_kernel<<<dim3((Cout * Cin) / 256), 256, 0, stream>>>(wgt, wt);
    conv3x3_wmma_fast<<<grid, 256, 0, stream>>>(xt, wt, out);
  } else {
    conv3x3_wmma_fallback<<<grid, 256, 0, stream>>>(x, wgt, out);
  }
}
